// MSASparseIndex_1228360646735
// MI455X (gfx1250) — matrix operand order repaired and checked by reading, not compiled, not run on hardware
//
#include <hip/hip_runtime.h>
#include <hip/hip_bf16.h>
#include <stdint.h>

// ---------------------------------------------------------------------------
// MI455X (gfx1250) fused retrieval pipeline.
// B=256, D=512, P=262144, H=4 heads (128 dims), TOP_K=16, scale = 1/(4*0.1)=2.5
// Prototype stream (512 MB) is moved by the Tensor Data Mover, double-buffered
// against the bf16 WMMA compute so DMA and math overlap.
// ---------------------------------------------------------------------------

#define B_ROWS   256
#define DIM      512
#define P_ROWS   262144
#define NWG      256              // workgroups for the fused score kernel
#define SLICE    (P_ROWS / NWG)   // 1024 prototypes per WG
#define SUBP     32               // prototypes per sub-tile (TDM tile)
#define NSUB     (SLICE / SUBP)   // 32 sub-tiles
#define SCALE_S  2.5f
#define NEG_BIG  -3.0e38f

typedef __attribute__((ext_vector_type(16))) __bf16 v16bf;
typedef __attribute__((ext_vector_type(8)))  float  v8f;
typedef unsigned int v4u __attribute__((ext_vector_type(4)));
typedef int          v4i __attribute__((ext_vector_type(4)));
typedef int          v8i __attribute__((ext_vector_type(8)));

// LDS carve for the fused kernel (dynamic, 233984 B <= 320KB/WGP):
#define L_PROTO0 0                //  32x512 fp32  = 65536
#define L_PROTO1 65536            //  32x512 fp32  = 65536
#define L_KN     131072           //  32x528 bf16  = 33792
#define L_SC     164864           // 256x33  fp32  = 33792
#define L_SS     198656           //  32x4   fp32  =   512
#define L_TS     199168           // 256x17  fp32  = 17408
#define L_TI     216576           // 256x17  i32   = 17408
#define SMEM2    233984

__device__ __forceinline__ uint16_t bf16_bits(float x) {
    union { __bf16 h; uint16_t u; } v;
    v.h = (__bf16)x;
    return v.u;
}

__device__ __forceinline__ v8f wmma_bf16(v16bf a, v16bf b, v8f c) {
    // v_wmma_f32_16x16x32_bf16 : D = A(16x32) * B(32x16) + C(16x16 f32)
    return __builtin_amdgcn_wmma_f32_16x16x32_bf16(
        false, a, false, b, (short)0, c, false, false);
}

// A-matrix (16x32 bf16) from row-major fp32 source.
// Lanes 0-15 = M, K {0..7,16..23}; lanes 16-31 = M, K {8..15,24..31}.
__device__ __forceinline__ v16bf loadA_f32(const float* base, int ldr, int row0, int k0, int lane) {
    int m   = row0 + (lane & 15);
    int sel = (lane >> 4) * 8;
    const float* r = base + (size_t)m * ldr;
    float4 x0 = *(const float4*)(r + k0 + sel);
    float4 x1 = *(const float4*)(r + k0 + sel + 4);
    float4 y0 = *(const float4*)(r + k0 + 16 + sel);
    float4 y1 = *(const float4*)(r + k0 + 16 + sel + 4);
    v16bf a;
    a[0]=(__bf16)x0.x; a[1]=(__bf16)x0.y; a[2]=(__bf16)x0.z; a[3]=(__bf16)x0.w;
    a[4]=(__bf16)x1.x; a[5]=(__bf16)x1.y; a[6]=(__bf16)x1.z; a[7]=(__bf16)x1.w;
    a[8]=(__bf16)y0.x; a[9]=(__bf16)y0.y; a[10]=(__bf16)y0.z; a[11]=(__bf16)y0.w;
    a[12]=(__bf16)y1.x; a[13]=(__bf16)y1.y; a[14]=(__bf16)y1.z; a[15]=(__bf16)y1.w;
    return a;
}

// A-matrix from row-major bf16 source.
__device__ __forceinline__ v16bf loadA_bf16(const uint16_t* base, int ldr, int row0, int k0, int lane) {
    int m   = row0 + (lane & 15);
    int sel = (lane >> 4) * 8;
    const uint16_t* r = base + (size_t)m * ldr;
    union { uint4 q[2]; v16bf v; } u;
    u.q[0] = *(const uint4*)(r + k0 + sel);
    u.q[1] = *(const uint4*)(r + k0 + 16 + sel);
    return u.v;
}

// B-matrix (32x16 bf16): column n of B = row n of a row-major bf16 source
// (computes X @ SRC^T). Lanes 0-15 = N, K {0..7,16..23}; lanes 16-31 = N, K {8..15,24..31}.
__device__ __forceinline__ v16bf loadB_bf16(const uint16_t* base, int ldr, int n0, int k0, int lane) {
    int n  = n0 + (lane & 15);
    int kk = k0 + (lane >> 4) * 8;
    const uint16_t* r = base + (size_t)n * ldr + kk;
    union { uint4 q[2]; v16bf v; } u;
    u.q[0] = *(const uint4*)r;
    u.q[1] = *(const uint4*)(r + 16);
    return u.v;
}

// Tensor Data Mover: DMA one 32x512 fp32 tile (row-major, stride 512) from
// global memory into LDS.  D# layout per CDNA5 ISA.
__device__ __forceinline__ void tdm_load_tile(uint32_t lds_addr, const float* gsrc) {
    uint64_t ga = (uint64_t)(uintptr_t)gsrc;
    v4u g0 = { 1u,                                   // count=1 (valid), user mode
               lds_addr,                             // lds_addr [63:32]
               (uint32_t)ga,                         // global_addr lo
               ((uint32_t)(ga >> 32) & 0x01FFFFFFu)  // global_addr [56:32]
               | 0x80000000u };                      // type=2 ("image")
    v8i g1 = { (int)(2u << 16),          // data_size=2 (4 bytes)
               (int)(512u << 16),        // tensor_dim0 = 512 (bits 79:48 lo)
               (int)((uint32_t)SUBP << 16), // tensor_dim1 = 32 (bits 111:80 lo)
               (int)(512u << 16),        // tile_dim0 = 512 (bits 127:112)
               SUBP,                     // tile_dim1 = 32 (bits 143:128)
               512,                      // tensor_dim0_stride = 512 (bits 207:160)
               0, 0 };
    v4i z4 = { 0, 0, 0, 0 };
#if defined(__clang_major__) && (__clang_major__ >= 23)
    v8i z8 = { 0, 0, 0, 0, 0, 0, 0, 0 };
    __builtin_amdgcn_tensor_load_to_lds(g0, g1, z4, z4, z8, 0);
#else
    __builtin_amdgcn_tensor_load_to_lds(g0, g1, z4, z4, 0);
#endif
}

// ---------------------------------------------------------------------------
// Kernel 0: fp32 -> bf16 conversion for the hot B operands (removes
// v_cvt-after-WMMA WAR hazards that force 4 co-exec NOPs per WMMA).
// ---------------------------------------------------------------------------
__global__ __launch_bounds__(256) void prep_kernel(const float* __restrict__ W,
                                                   uint16_t* __restrict__ Wb, int n) {
    for (int i = blockIdx.x * 256 + threadIdx.x; i < n; i += gridDim.x * 256)
        Wb[i] = bf16_bits(W[i]);
}

// ---------------------------------------------------------------------------
// Kernel 1: qn = head_normalize(h @ W_QR^T) as bf16.  One wave per
// (16-row-tile x 128-col head) so the per-head L2 norm is wave-local.
// ---------------------------------------------------------------------------
__global__ __launch_bounds__(32) void qn_kernel(const float* __restrict__ h,
                                                const uint16_t* __restrict__ WQRb,
                                                uint16_t* __restrict__ qn) {
    const int lane = threadIdx.x;
    const int rt   = blockIdx.x & 15;   // 16 row tiles over B=256
    const int head = blockIdx.x >> 4;   // 4 heads
    const int row0 = rt * 16;
    const int c0   = head * 128;

    v8f acc[8];
    v8f zero8 = {};
#pragma unroll
    for (int t = 0; t < 8; ++t) acc[t] = zero8;

    for (int k0 = 0; k0 < DIM; k0 += 32) {
        v16bf a = loadA_f32(h, DIM, row0, k0, lane);
#pragma unroll
        for (int t = 0; t < 8; ++t) {
            v16bf b = loadB_bf16(WQRb, DIM, c0 + t * 16, k0, lane);
            acc[t] = wmma_bf16(a, b, acc[t]);
        }
    }
    float inv[8];
#pragma unroll
    for (int j = 0; j < 8; ++j) {
        float s = 0.f;
#pragma unroll
        for (int t = 0; t < 8; ++t) { float v = acc[t][j]; s += v * v; }
        s += __shfl_xor(s, 1, 32);
        s += __shfl_xor(s, 2, 32);
        s += __shfl_xor(s, 4, 32);
        s += __shfl_xor(s, 8, 32);
        inv[j] = 1.0f / fmaxf(sqrtf(s), 1e-12f);
    }
#pragma unroll
    for (int t = 0; t < 8; ++t)
#pragma unroll
        for (int j = 0; j < 8; ++j) {
            int r = row0 + j + 8 * (lane >> 4);
            int c = c0 + t * 16 + (lane & 15);
            qn[(size_t)r * DIM + c] = bf16_bits(acc[t][j] * inv[j]);
        }
}

// ---------------------------------------------------------------------------
// Kernel 2: fused  kr = proto @ W_KR^T -> head-normalize -> scores = qn@kn^T
//           -> running per-row top-16.  Prototypes streamed once via TDM,
//           double-buffered in LDS against the WMMA compute.
// ---------------------------------------------------------------------------
__global__ __launch_bounds__(256) void score_topk_kernel(
        const float* __restrict__ prototypes,
        const uint16_t* __restrict__ WKRb,
        const uint16_t* __restrict__ qn,
        float* __restrict__ cand_s,
        int*   __restrict__ cand_i) {
    extern __shared__ char smem[];
    uint16_t* lds_kn = (uint16_t*)(smem + L_KN);
    float*    lds_sc = (float*)(smem + L_SC);
    float*    lds_ss = (float*)(smem + L_SS);
    float*    lds_ts = (float*)(smem + L_TS);
    int*      lds_ti = (int*)(smem + L_TI);

    const int t    = threadIdx.x;
    const int lane = t & 31;
    const int wave = t >> 5;           // 0..7
    const v8f zero8 = {};

    const uint32_t lds_base = (uint32_t)(uintptr_t)smem;
    const float*   gslice   = prototypes + (size_t)blockIdx.x * SLICE * DIM;

    float thr = NEG_BIG;
#pragma unroll
    for (int k = 0; k < 16; ++k) { lds_ts[t * 17 + k] = NEG_BIG; lds_ti[t * 17 + k] = 0; }

    // prologue: DMA sub-tile 0 into buffer 0
    if (wave == 0)
        tdm_load_tile(lds_base + L_PROTO0, gslice);

    for (int sub = 0; sub < NSUB; ++sub) {
        const int   pbase = blockIdx.x * SLICE + sub * SUBP;
        const int   cur   = sub & 1;
        const float* lds_proto = (const float*)(smem + (cur ? L_PROTO1 : L_PROTO0));

        if (wave == 0)
            __builtin_amdgcn_s_wait_tensorcnt(0);   // current buffer landed
        if (t < SUBP * 4) lds_ss[t] = 0.f;          // zero head sum-squares
        __syncthreads();

        // kick off DMA for the next sub-tile into the other buffer
        if (wave == 0 && sub + 1 < NSUB)
            tdm_load_tile(lds_base + (cur ? L_PROTO0 : L_PROTO1),
                          gslice + (size_t)(sub + 1) * SUBP * DIM);

        // ---- step 1: kr tile. wave w -> 2 row-tiles x 4 col-tiles
        //      (cols w*64 .. w*64+63, entirely inside head w>>1) ----
        const int c0   = wave * 64;
        const int head = wave >> 1;
        v8f acc[2][4];
#pragma unroll
        for (int rt = 0; rt < 2; ++rt)
#pragma unroll
            for (int i = 0; i < 4; ++i) acc[rt][i] = zero8;

        for (int kk = 0; kk < 16; ++kk) {
            const int k0 = kk * 32;
            v16bf A0 = loadA_f32(lds_proto, DIM, 0,  k0, lane);
            v16bf A1 = loadA_f32(lds_proto, DIM, 16, k0, lane);
#pragma unroll
            for (int i = 0; i < 4; ++i) {
                v16bf Bm = loadB_bf16(WKRb, DIM, c0 + i * 16, k0, lane);
                acc[0][i] = wmma_bf16(A0, Bm, acc[0][i]);
                acc[1][i] = wmma_bf16(A1, Bm, acc[1][i]);
            }
        }
        // partial per-head sum of squares -> LDS atomics
#pragma unroll
        for (int rt = 0; rt < 2; ++rt)
#pragma unroll
            for (int j = 0; j < 8; ++j) {
                float s = 0.f;
#pragma unroll
                for (int i = 0; i < 4; ++i) { float v = acc[rt][i][j]; s += v * v; }
                s += __shfl_xor(s, 1, 32);
                s += __shfl_xor(s, 2, 32);
                s += __shfl_xor(s, 4, 32);
                s += __shfl_xor(s, 8, 32);
                if ((lane & 15) == 0) {
                    int row = rt * 16 + j + 8 * (lane >> 4);
                    atomicAdd(&lds_ss[row * 4 + head], s);
                }
            }
        __syncthreads();

        // normalize and write kn (bf16) to LDS
#pragma unroll
        for (int rt = 0; rt < 2; ++rt)
#pragma unroll
            for (int j = 0; j < 8; ++j) {
                int row = rt * 16 + j + 8 * (lane >> 4);
                float invn = 1.0f / fmaxf(sqrtf(lds_ss[row * 4 + head]), 1e-12f);
#pragma unroll
                for (int i = 0; i < 4; ++i) {
                    int col = c0 + i * 16 + (lane & 15);
                    lds_kn[row * 528 + col] = bf16_bits(acc[rt][i][j] * invn);
                }
            }
        __syncthreads();

        // ---- step 2: scores = qn(256x512) @ kn^T(512x32); wave owns 32 batch
        //      rows x all 32 prototypes ----
        v8f sa[2][2];
#pragma unroll
        for (int ii = 0; ii < 2; ++ii)
#pragma unroll
            for (int pt = 0; pt < 2; ++pt) sa[ii][pt] = zero8;

        for (int kk = 0; kk < 16; ++kk) {
            const int k0 = kk * 32;
            v16bf a0 = loadA_bf16(qn, DIM, wave * 32,      k0, lane);
            v16bf a1 = loadA_bf16(qn, DIM, wave * 32 + 16, k0, lane);
#pragma unroll
            for (int pt = 0; pt < 2; ++pt) {
                v16bf b = loadB_bf16(lds_kn, 528, pt * 16, k0, lane);
                sa[0][pt] = wmma_bf16(a0, b, sa[0][pt]);
                sa[1][pt] = wmma_bf16(a1, b, sa[1][pt]);
            }
        }
#pragma unroll
        for (int ii = 0; ii < 2; ++ii)
#pragma unroll
            for (int pt = 0; pt < 2; ++pt)
#pragma unroll
                for (int j = 0; j < 8; ++j) {
                    int b = wave * 32 + ii * 16 + j + 8 * (lane >> 4);
                    int p = pt * 16 + (lane & 15);
                    lds_sc[b * 33 + p] = sa[ii][pt][j] * SCALE_S;
                }
        __syncthreads();

        // ---- step 3: per-thread (= per batch row) top-16 update ----
        for (int j = 0; j < SUBP; ++j) {
            float s = lds_sc[t * 33 + j];
            if (s > thr) {
                float mv = lds_ts[t * 17]; int am = 0;
#pragma unroll
                for (int k = 1; k < 16; ++k) {
                    float v = lds_ts[t * 17 + k];
                    if (v < mv) { mv = v; am = k; }
                }
                lds_ts[t * 17 + am] = s;
                lds_ti[t * 17 + am] = pbase + j;
                mv = lds_ts[t * 17];
#pragma unroll
                for (int k = 1; k < 16; ++k) mv = fminf(mv, lds_ts[t * 17 + k]);
                thr = mv;
            }
        }
        __syncthreads();
    }

    // emit this slice's top-16 candidates per batch row
    {
        size_t base = ((size_t)blockIdx.x * B_ROWS + t) * 16;
#pragma unroll
        for (int k = 0; k < 16; ++k) {
            cand_s[base + k] = lds_ts[t * 17 + k];
            cand_i[base + k] = lds_ti[t * 17 + k];
        }
    }
}

// ---------------------------------------------------------------------------
// Kernel 3: per batch row, exact top-16 of NWG*16 candidates, softmax,
// weighted gather of raw prototype rows -> retrieved[256x512].
// ---------------------------------------------------------------------------
__global__ __launch_bounds__(256) void merge_kernel(const float* __restrict__ cand_s,
                                                    const int*   __restrict__ cand_i,
                                                    const float* __restrict__ prototypes,
                                                    float* __restrict__ retrieved) {
    __shared__ float ss[NWG * 16];
    __shared__ int   si[NWG * 16];
    __shared__ float rv[256];
    __shared__ int   ri[256];
    __shared__ float selw[16];
    __shared__ int   seli[16];

    const int b = blockIdx.x, t = threadIdx.x;
#pragma unroll
    for (int i = 0; i < 16; ++i) {
        int e = t * 16 + i;                          // thread t owns slice wg=t
        ss[e] = cand_s[((size_t)t * B_ROWS + b) * 16 + i];
        si[e] = cand_i[((size_t)t * B_ROWS + b) * 16 + i];
    }
    __syncthreads();

    for (int r = 0; r < 16; ++r) {
        float bv = NEG_BIG; int be = t * 16;
#pragma unroll
        for (int i = 0; i < 16; ++i) {
            int e = t * 16 + i; float v = ss[e];
            if (v > bv) { bv = v; be = e; }
        }
        rv[t] = bv; ri[t] = be;
        __syncthreads();
        for (int off = 128; off > 0; off >>= 1) {
            if (t < off && rv[t + off] > rv[t]) { rv[t] = rv[t + off]; ri[t] = ri[t + off]; }
            __syncthreads();
        }
        if (t == 0) { selw[r] = rv[0]; seli[r] = si[ri[0]]; ss[ri[0]] = NEG_BIG; }
        __syncthreads();
    }

    if (t == 0) {
        float mx = selw[0];
#pragma unroll
        for (int k = 1; k < 16; ++k) mx = fmaxf(mx, selw[k]);
        float sum = 0.f;
#pragma unroll
        for (int k = 0; k < 16; ++k) { float w = __expf(selw[k] - mx); selw[k] = w; sum += w; }
        float inv = 1.0f / sum;
#pragma unroll
        for (int k = 0; k < 16; ++k) selw[k] *= inv;
    }
    __syncthreads();

#pragma unroll
    for (int half = 0; half < 2; ++half) {
        int d = t + half * 256;
        float acc = 0.f;
#pragma unroll
        for (int k = 0; k < 16; ++k)
            acc += selw[k] * prototypes[(size_t)seli[k] * DIM + d];
        retrieved[(size_t)b * DIM + d] = acc;
    }
}

// ---------------------------------------------------------------------------
// Kernel 4: out = retrieved @ W_out^T (bf16 WMMA, fp32 out). One wave per tile.
// ---------------------------------------------------------------------------
__global__ __launch_bounds__(32) void out_kernel(const float* __restrict__ R,
                                                 const uint16_t* __restrict__ WOutb,
                                                 float* __restrict__ out) {
    const int lane = threadIdx.x;
    const int rt = blockIdx.x & 15;
    const int ct = blockIdx.x >> 4;        // 32 col tiles
    const int row0 = rt * 16, c0 = ct * 16;
    v8f acc = {};
    for (int kk = 0; kk < 16; ++kk) {
        v16bf a = loadA_f32(R, DIM, row0, kk * 32, lane);
        v16bf b = loadB_bf16(WOutb, DIM, c0, kk * 32, lane);
        acc = wmma_bf16(a, b, acc);
    }
#pragma unroll
    for (int j = 0; j < 8; ++j) {
        int r = row0 + j + 8 * (lane >> 4);
        int c = c0 + (lane & 15);
        out[(size_t)r * DIM + c] = acc[j];
    }
}

// ---------------------------------------------------------------------------
extern "C" void kernel_launch(void* const* d_in, const int* in_sizes, int n_in,
                              void* d_out, int out_size, void* d_ws, size_t ws_size,
                              hipStream_t stream) {
    (void)in_sizes; (void)n_in; (void)out_size; (void)ws_size;
    const float* h          = (const float*)d_in[0];
    const float* prototypes = (const float*)d_in[1];
    const float* W_QR       = (const float*)d_in[2];
    const float* W_KR       = (const float*)d_in[3];
    const float* W_out      = (const float*)d_in[4];
    float* out = (float*)d_out;

    char* ws = (char*)d_ws;
    uint16_t* qn     = (uint16_t*)ws;                               // 256 KB
    uint16_t* WKRb   = (uint16_t*)(ws + 262144);                    // 512 KB
    uint16_t* WQRb   = (uint16_t*)(ws + 262144 + 524288);           // 512 KB
    uint16_t* WOutb  = (uint16_t*)(ws + 262144 + 2 * 524288);       // 512 KB
    float*    cand_s = (float*)(ws + 262144 + 3 * 524288);          // 4 MB
    int*      cand_i = (int*)(ws + 262144 + 3 * 524288 + 4194304);  // 4 MB
    float*    retr   = (float*)(ws + 262144 + 3 * 524288 + 8388608);// 512 KB

    hipFuncSetAttribute((const void*)score_topk_kernel,
                        hipFuncAttributeMaxDynamicSharedMemorySize, SMEM2);

    prep_kernel<<<256, 256, 0, stream>>>(W_KR, WKRb, DIM * DIM);
    prep_kernel<<<256, 256, 0, stream>>>(W_QR, WQRb, DIM * DIM);
    prep_kernel<<<256, 256, 0, stream>>>(W_out, WOutb, DIM * DIM);
    qn_kernel<<<64, 32, 0, stream>>>(h, WQRb, qn);
    score_topk_kernel<<<NWG, 256, SMEM2, stream>>>(prototypes, WKRb, qn, cand_s, cand_i);
    merge_kernel<<<B_ROWS, 256, 0, stream>>>(cand_s, cand_i, prototypes, retr);
    out_kernel<<<512, 32, 0, stream>>>(retr, WOutb, out);
}
